// SelfAttention_24120536334752
// MI455X (gfx1250) — matrix operand order repaired and checked by reading, not compiled, not run on hardware
//
#include <hip/hip_runtime.h>

// ---------------------------------------------------------------------------
// Problem constants (from reference): E=1024, H=16, S=2048, N=2, DH=64
// ---------------------------------------------------------------------------
#define E_DIM  1024
#define H_DIM  16
#define S_DIM  2048
#define N_DIM  2
#define DH_DIM 64

typedef _Float16 v16h __attribute__((ext_vector_type(16)));
typedef _Float16 h8   __attribute__((ext_vector_type(8)));
typedef _Float16 h4   __attribute__((ext_vector_type(4)));
typedef float    v8f  __attribute__((ext_vector_type(8)));
typedef unsigned int u32x4 __attribute__((ext_vector_type(4)));
typedef int          i32x8 __attribute__((ext_vector_type(8)));
typedef int          i32x4 __attribute__((ext_vector_type(4)));

static __device__ __forceinline__ v16h comb16(h8 lo, h8 hi) {
  return __builtin_shufflevector(lo, hi, 0,1,2,3,4,5,6,7,8,9,10,11,12,13,14,15);
}

static __device__ __forceinline__ v8f wmma_f16(v16h a, v16h b, v8f c) {
  // D(f32 16x16) = A(f16 16x32) * B(f16 32x16) + C
  return __builtin_amdgcn_wmma_f32_16x16x32_f16(false, a, false, b, (short)0, c,
                                                false, false);
}

// A fragment (16x32 f16, M x K). Layout: lane m = L%16, hf = L/16.
// element e -> K = hf*16 + e (32B contiguous; this code's order, same as B, not ISA table order)
static __device__ __forceinline__ v16h load_afrag(const _Float16* rowptr, int hf) {
  h8 lo = *(const h8*)(rowptr + hf * 16);
  h8 hi = *(const h8*)(rowptr + hf * 16 + 8);
  return comb16(lo, hi);
}

// B fragment (32x16 f16, K x N). Layout: lane n = L%16, hf = L/16,
// element e -> K = hf*16 + e (this code's order, same as A). p points at this lane's 16 contiguous K values.
static __device__ __forceinline__ v16h load_bfrag(const _Float16* p) {
  h8 lo = *(const h8*)p;
  h8 hi = *(const h8*)(p + 8);
  return comb16(lo, hi);
}

// ---------------------------------------------------------------------------
// Tensor Data Mover: DMA a 64(rows) x 64(f16) tile of a row-major
// [rows_total x kdim] f16 matrix (gsrc = tile origin) into LDS, inserting
// 4 DWORDs (8 f16) of padding after every 32 DWORDs (one 128B row) so the
// LDS image has row pitch 72 f16 — matching Bs[64][72].
// D# layout per CDNA5 ISA 8.3/8.4; groups 2/3 zero (2-D tensor).
// ---------------------------------------------------------------------------
static __device__ __forceinline__ void tdm_load_tile_f16(
    const _Float16* gsrc, int rows_total, int kdim, void* lds_dst) {
  unsigned long long ga = (unsigned long long)(const void*)gsrc;
  unsigned int lds = (unsigned int)(size_t)lds_dst;  // low 32 bits = LDS offset
  u32x4 g0;
  g0[0] = 1u;                                           // count=1, no gather
  g0[1] = lds;                                          // lds_addr
  g0[2] = (unsigned int)ga;                             // global_addr[31:0]
  g0[3] = ((unsigned int)(ga >> 32) & 0x1FFFFFFu)       // global_addr[56:32]
          | (2u << 30);                                 // type=2 ("image")
  i32x8 g1;
  g1[0] = (1 << 16)      // data_size = 2 bytes
        | (1 << 20)      // pad_enable
        | (4 << 22)      // pad_interval: 2^(4+1)=32 DWORDs (one 128B row)
        | (3 << 25);     // pad_amount: 4 DWORDs (8 f16)
  g1[1] = (int)((unsigned)kdim << 16);                  // tensor_dim0 lo16
  g1[2] = (int)(((unsigned)rows_total << 16) |          // tensor_dim1 lo16
                ((unsigned)kdim >> 16));                // tensor_dim0 hi16
  g1[3] = (int)((64u << 16) |                           // tile_dim0 = 64
                ((unsigned)rows_total >> 16));          // tensor_dim1 hi16
  g1[4] = 64;                                           // tile_dim1 = 64 rows
  g1[5] = kdim;                                         // dim0_stride lo32
  g1[6] = 0;                                            // dim0_stride hi16 etc
  g1[7] = 0;
  i32x4 z4 = {0, 0, 0, 0};
#if defined(__clang_major__) && __clang_major__ >= 23
  i32x8 z8 = {0, 0, 0, 0, 0, 0, 0, 0};
  __builtin_amdgcn_tensor_load_to_lds(g0, g1, z4, z4, z8, 0);
#else
  __builtin_amdgcn_tensor_load_to_lds(g0, g1, z4, z4, 0);
#endif
}

// ---------------------------------------------------------------------------
// Kernel 0a: elementwise f32 -> f16 (seq).
// ---------------------------------------------------------------------------
__global__ __launch_bounds__(256)
void cvt_f16_kernel(const float* __restrict__ src, _Float16* __restrict__ dst,
                    int n4) {
  int i = blockIdx.x * 256 + threadIdx.x;
  if (i < n4) {
    float4 f = ((const float4*)src)[i];
    h4 h;
    h[0] = (_Float16)f.x; h[1] = (_Float16)f.y;
    h[2] = (_Float16)f.z; h[3] = (_Float16)f.w;
    ((h4*)dst)[i] = h;
  }
}

// ---------------------------------------------------------------------------
// Kernel 0b: transpose + convert.  W f32 [K x Nc] -> Wt f16 [Nc x K].
// 32x32 LDS tile, block (32,8).
// ---------------------------------------------------------------------------
__global__ __launch_bounds__(256)
void transpose_cvt_kernel(const float* __restrict__ W, _Float16* __restrict__ Wt,
                          int K, int Nc) {
  __shared__ _Float16 T[32][33];
  const int j0 = blockIdx.x * 32;
  const int k0 = blockIdx.y * 32;
  const int tx = threadIdx.x;
  const int ty = threadIdx.y;
#pragma unroll
  for (int i = 0; i < 4; ++i) {
    int k = ty + i * 8;
    T[k][tx] = (_Float16)W[(size_t)(k0 + k) * Nc + j0 + tx];
  }
  __syncthreads();
#pragma unroll
  for (int i = 0; i < 4; ++i) {
    int j = ty + i * 8;
    Wt[(size_t)(j0 + j) * K + k0 + tx] = T[tx][j];
  }
}

// ---------------------------------------------------------------------------
// Shared GEMM core: C[64x64] tile of A[M x K](f16,row-major) @ Bt[Nc x K]
// (f16, pre-transposed: row j = column j of B). BK = 64, 128 threads, 4 waves.
// A tile staged with vector loads; B tile staged by the Tensor Data Mover,
// double-buffered so the DMA overlaps the WMMAs (wave 0 issues + waits
// TENSORcnt; barriers publish the tile to the other waves).
// ---------------------------------------------------------------------------
#define GEMM_CORE(Ah, Bth, Kdim, NcRows)                                       \
  __shared__ _Float16 As[64][72];                                              \
  __shared__ _Float16 Bs[2][64][72];                                           \
  const int tid  = threadIdx.x;                                                \
  const int w    = tid >> 5;                                                   \
  const int lane = tid & 31;                                                   \
  const int nl   = lane & 15;                                                  \
  const int hf   = lane >> 4;                                                  \
  const int m0   = blockIdx.y * 64;                                            \
  const int n0   = blockIdx.x * 64;                                            \
  v8f acc[4];                                                                  \
  _Pragma("unroll") for (int dt = 0; dt < 4; ++dt) { v8f z = {}; acc[dt] = z; }\
  if (w == 0)                                                                  \
    tdm_load_tile_f16((Bth) + (size_t)n0 * (Kdim), (NcRows), (Kdim),           \
                      &Bs[0][0][0]);                                           \
  int buf = 0;                                                                 \
  for (int k0 = 0; k0 < (Kdim); k0 += 64, buf ^= 1) {                          \
    _Pragma("unroll") for (int i = 0; i < 4; ++i) {                            \
      int c = tid + i * 128;                                                   \
      int r = c >> 3, ko = (c & 7) * 8;                                        \
      *(uint4*)&As[r][ko] =                                                    \
          *(const uint4*)((Ah) + (size_t)(m0 + r) * (Kdim) + k0 + ko);         \
    }                                                                          \
    if (w == 0) __builtin_amdgcn_s_wait_tensorcnt((short)0);                   \
    __syncthreads();                                                           \
    if (w == 0 && k0 + 64 < (Kdim))                                            \
      tdm_load_tile_f16((Bth) + (size_t)n0 * (Kdim) + k0 + 64, (NcRows),       \
                        (Kdim), &Bs[buf ^ 1][0][0]);                           \
    v16h a0 = load_afrag(&As[w * 16 + nl][0], hf);                             \
    v16h a1 = load_afrag(&As[w * 16 + nl][32], hf);                            \
    _Pragma("unroll") for (int dt = 0; dt < 4; ++dt) {                         \
      v16h b0 = load_bfrag(&Bs[buf][dt * 16 + nl][hf * 16]);                   \
      v16h b1 = load_bfrag(&Bs[buf][dt * 16 + nl][32 + hf * 16]);              \
      acc[dt] = wmma_f16(a0, b0, acc[dt]);                                     \
      acc[dt] = wmma_f16(a1, b1, acc[dt]);                                     \
    }                                                                          \
    __syncthreads();                                                           \
  }

// ---------------------------------------------------------------------------
// Kernel 1: QKV projection. C[4096 x 3072] = seq_h @ Wqkv_t^T + b.
// Scatter f16 to Q[N,H,S,DH], K[N,H,S,DH], Vt[N,H,DH,S].
// which / h are uniform per block (n0 multiple of 64).
// ---------------------------------------------------------------------------
__global__ __launch_bounds__(128)
void qkv_gemm_kernel(const _Float16* __restrict__ Ah,
                     const _Float16* __restrict__ Bth,
                     const float* __restrict__ bias,
                     _Float16* __restrict__ Qo, _Float16* __restrict__ Ko,
                     _Float16* __restrict__ Vto) {
  GEMM_CORE(Ah, Bth, E_DIM, 3 * E_DIM)

  const int which = n0 >> 10;                 // 0=q 1=k 2=v (uniform)
  const int h     = (n0 & (E_DIM - 1)) >> 6;  // head (uniform)

  if (which == 2) {
    // V transposed: [N,H,DH,S]
#pragma unroll
    for (int dt = 0; dt < 4; ++dt) {
      int dd = dt * 16 + nl;
      float bv = bias[n0 + dd];
#pragma unroll
      for (int r = 0; r < 8; ++r) {
        int t = m0 + w * 16 + r + 8 * hf;
        int nb = t >> 11, s = t & (S_DIM - 1);
        Vto[(((size_t)nb * H_DIM + h) * DH_DIM + dd) * S_DIM + s] =
            (_Float16)(acc[dt][r] + bv);
      }
    }
  } else {
    _Float16* dst = (which == 0) ? Qo : Ko;  // [N,H,S,DH]
#pragma unroll
    for (int dt = 0; dt < 4; ++dt) {
      int dd = dt * 16 + nl;
      float bv = bias[n0 + dd];
#pragma unroll
      for (int r = 0; r < 8; ++r) {
        int t = m0 + w * 16 + r + 8 * hf;
        int nb = t >> 11, s = t & (S_DIM - 1);
        dst[(((size_t)nb * H_DIM + h) * S_DIM + s) * DH_DIM + dd] =
            (_Float16)(acc[dt][r] + bv);
      }
    }
  }
}

// ---------------------------------------------------------------------------
// Kernel 2: causal flash attention.
// Grid: (S/64, H, N). Block: 128 threads (4 waves). Each wave owns 16 query
// rows; streams kv in steps of 32; O (16x64) stays in f32 accumulators.
// Output written f16 [N,S,E] (feeds final projection directly).
// ---------------------------------------------------------------------------
__global__ __launch_bounds__(128)
void attn_kernel(const _Float16* __restrict__ Q, const _Float16* __restrict__ Kp,
                 const _Float16* __restrict__ Vt, _Float16* __restrict__ O) {
  const int w    = threadIdx.x >> 5;
  const int lane = threadIdx.x & 31;
  const int nl   = lane & 15;
  const int hf   = lane >> 4;
  const int h    = blockIdx.y;
  const int nb   = blockIdx.z;
  const int qb   = blockIdx.x * 64 + w * 16;

  const _Float16* Qh = Q  + ((size_t)nb * H_DIM + h) * S_DIM * DH_DIM;
  const _Float16* Kh = Kp + ((size_t)nb * H_DIM + h) * S_DIM * DH_DIM;
  const _Float16* Vh = Vt + ((size_t)nb * H_DIM + h) * DH_DIM * S_DIM;

  __shared__ _Float16 Pb[4][16][40];  // per-wave P transpose buffer

  // Q A-fragments (DH=64 -> two 16x32 fragments), loaded once.
  const _Float16* qrow = Qh + (size_t)(qb + nl) * DH_DIM;
  v16h aQ0 = load_afrag(qrow, hf);
  v16h aQ1 = load_afrag(qrow + 32, hf);

  v8f accO[4];
#pragma unroll
  for (int dt = 0; dt < 4; ++dt) { v8f z = {}; accO[dt] = z; }
  float mrow[8], lrow[8];
#pragma unroll
  for (int r = 0; r < 8; ++r) { mrow[r] = -3.0e38f; lrow[r] = 0.0f; }

  const float scale = 0.125f;  // 1/sqrt(64)

  for (int s0 = 0; s0 < qb + 16; s0 += 32) {
    // Prefetch next kv tile (K rows and Vt rows) into cache while we compute.
    __builtin_prefetch(Kh + (size_t)(s0 + 32 + nl) * DH_DIM + hf * 32, 0, 1);
    __builtin_prefetch(Vh + (size_t)(nl + hf * 16) * S_DIM + s0 + 32, 0, 1);

    // ---- S tile = Q @ K^T (16q x 32kv) ----
    v8f sacc[2];
#pragma unroll
    for (int sub = 0; sub < 2; ++sub) {
      const _Float16* krow = Kh + (size_t)(s0 + sub * 16 + nl) * DH_DIM;
      v8f c = {};
      c = wmma_f16(aQ0, load_bfrag(krow + hf * 16), c);
      c = wmma_f16(aQ1, load_bfrag(krow + 32 + hf * 16), c);
      sacc[sub] = c;
    }

    // ---- scale + causal mask ----
#pragma unroll
    for (int r = 0; r < 8; ++r) {
      int qi = qb + r + 8 * hf;
#pragma unroll
      for (int sub = 0; sub < 2; ++sub) {
        int kv = s0 + sub * 16 + nl;
        float v = sacc[sub][r] * scale;
        sacc[sub][r] = (kv > qi) ? -3.0e38f : v;
      }
    }

    // ---- row max over 32 kv (2 subtiles + 16-lane half reduction) ----
    float tmax[8];
#pragma unroll
    for (int r = 0; r < 8; ++r) tmax[r] = fmaxf(sacc[0][r], sacc[1][r]);
#pragma unroll
    for (int off = 1; off < 16; off <<= 1) {
#pragma unroll
      for (int r = 0; r < 8; ++r)
        tmax[r] = fmaxf(tmax[r], __shfl_xor(tmax[r], off, 32));
    }

    // ---- online softmax update ----
    float fac[8], tsum[8];
#pragma unroll
    for (int r = 0; r < 8; ++r) {
      float mnew = fmaxf(mrow[r], tmax[r]);
      fac[r]  = __expf(mrow[r] - mnew);
      mrow[r] = mnew;
      float p0 = __expf(sacc[0][r] - mnew);
      float p1 = __expf(sacc[1][r] - mnew);
      sacc[0][r] = p0;
      sacc[1][r] = p1;
      tsum[r] = p0 + p1;
    }
#pragma unroll
    for (int off = 1; off < 16; off <<= 1) {
#pragma unroll
      for (int r = 0; r < 8; ++r) tsum[r] += __shfl_xor(tsum[r], off, 32);
    }
#pragma unroll
    for (int r = 0; r < 8; ++r) lrow[r] = lrow[r] * fac[r] + tsum[r];
#pragma unroll
    for (int dt = 0; dt < 4; ++dt)
#pragma unroll
      for (int r = 0; r < 8; ++r) accO[dt][r] *= fac[r];

    // ---- transpose P (C layout -> A layout) through per-wave LDS ----
#pragma unroll
    for (int r = 0; r < 8; ++r) {
      int m = r + 8 * hf;
      Pb[w][m][nl]      = (_Float16)sacc[0][r];
      Pb[w][m][nl + 16] = (_Float16)sacc[1][r];
    }
    h8 plo = *(const h8*)&Pb[w][nl][hf * 16];
    h8 phi = *(const h8*)&Pb[w][nl][hf * 16 + 8];
    v16h aP = comb16(plo, phi);

    // ---- O += P @ V  (V stored transposed [DH][S]) ----
#pragma unroll
    for (int dt = 0; dt < 4; ++dt) {
      const _Float16* vrow = Vh + (size_t)(dt * 16 + nl) * S_DIM + s0 + hf * 16;
      accO[dt] = wmma_f16(aP, load_bfrag(vrow), accO[dt]);
    }
  }

  // ---- epilogue: O / l, write f16 [N,S,E] ----
#pragma unroll
  for (int r = 0; r < 8; ++r) {
    float inv = 1.0f / lrow[r];
    int qi = qb + r + 8 * hf;
    _Float16* orow = O + ((size_t)nb * S_DIM + qi) * E_DIM + h * DH_DIM;
#pragma unroll
    for (int dt = 0; dt < 4; ++dt)
      orow[dt * 16 + nl] = (_Float16)(accO[dt][r] * inv);
  }
}

// ---------------------------------------------------------------------------
// Kernel 3: output projection. out[4096 x 1024] = O @ Wo_t^T + b_o (f32 out).
// ---------------------------------------------------------------------------
__global__ __launch_bounds__(128)
void out_gemm_kernel(const _Float16* __restrict__ Ah,
                     const _Float16* __restrict__ Bth,
                     const float* __restrict__ bias, float* __restrict__ Cout) {
  GEMM_CORE(Ah, Bth, E_DIM, E_DIM)
#pragma unroll
  for (int dt = 0; dt < 4; ++dt) {
    int j = n0 + dt * 16 + nl;
    float bv = bias[j];
#pragma unroll
    for (int r = 0; r < 8; ++r) {
      int t = m0 + w * 16 + r + 8 * hf;
      Cout[(size_t)t * E_DIM + j] = acc[dt][r] + bv;
    }
  }
}

// ---------------------------------------------------------------------------
// Launch: seq, mask, W_qkv, b_qkv, W_o, b_o  ->  out [N,S,E] f32
// Workspace layout (bytes):
//   seq_h  f16 [4096,1024]     : 8,388,608
//   Wqkv_t f16 [3072,1024]     : 6,291,456
//   Wo_t   f16 [1024,1024]     : 2,097,152
//   Q      f16 [N,H,S,DH]      : 8,388,608
//   K      f16 [N,H,S,DH]      : 8,388,608
//   Vt     f16 [N,H,DH,S]      : 8,388,608
//   O      f16 [N,S,E]         : 8,388,608     total ~50.3 MB
// ---------------------------------------------------------------------------
extern "C" void kernel_launch(void* const* d_in, const int* in_sizes, int n_in,
                              void* d_out, int out_size, void* d_ws, size_t ws_size,
                              hipStream_t stream) {
  (void)in_sizes; (void)n_in; (void)out_size; (void)ws_size;
  const float* seq  = (const float*)d_in[0];
  // d_in[1] = mask (causal; applied analytically in-kernel)
  const float* Wqkv = (const float*)d_in[2];
  const float* bqkv = (const float*)d_in[3];
  const float* Wo   = (const float*)d_in[4];
  const float* bo   = (const float*)d_in[5];
  float* out = (float*)d_out;

  char* ws = (char*)d_ws;
  size_t off = 0;
  _Float16* seq_h  = (_Float16*)(ws + off); off += (size_t)4096 * 1024 * 2;
  _Float16* Wqkv_t = (_Float16*)(ws + off); off += (size_t)3072 * 1024 * 2;
  _Float16* Wo_t   = (_Float16*)(ws + off); off += (size_t)1024 * 1024 * 2;
  _Float16* Qb     = (_Float16*)(ws + off); off += (size_t)4096 * 1024 * 2;
  _Float16* Kb     = (_Float16*)(ws + off); off += (size_t)4096 * 1024 * 2;
  _Float16* Vt     = (_Float16*)(ws + off); off += (size_t)4096 * 1024 * 2;
  _Float16* Ob     = (_Float16*)(ws + off);

  // 0) One-time precision/layout conversion.
  cvt_f16_kernel<<<4096, 256, 0, stream>>>(seq, seq_h, 4096 * 1024 / 4);
  transpose_cvt_kernel<<<dim3(96, 32), dim3(32, 8), 0, stream>>>(
      Wqkv, Wqkv_t, E_DIM, 3 * E_DIM);
  transpose_cvt_kernel<<<dim3(32, 32), dim3(32, 8), 0, stream>>>(
      Wo, Wo_t, E_DIM, E_DIM);

  // 1) QKV projection: M=4096, N=3072 -> grid (48, 64)
  qkv_gemm_kernel<<<dim3(48, 64), 128, 0, stream>>>(seq_h, Wqkv_t, bqkv,
                                                    Qb, Kb, Vt);
  // 2) Flash attention: grid (S/64, H, N)
  attn_kernel<<<dim3(S_DIM / 64, H_DIM, N_DIM), 128, 0, stream>>>(Qb, Kb, Vt, Ob);
  // 3) Output projection: M=4096, N=1024 -> grid (16, 64)
  out_gemm_kernel<<<dim3(16, 64), 128, 0, stream>>>(Ob, Wo_t, bo, out);
}
